// MoleculeGCN_50826642980949
// MI455X (gfx1250) — compile-verified, matrix operand order checked by reading, not run on hardware
//
#include <hip/hip_runtime.h>
#include <hip/hip_bf16.h>
#include <float.h>

#define N_NODES   100000
#define N_EDGES   800000
#define N_GRAPHS  512
#define ATOM_DIM  32
#define BOND_DIM  16
#define EMB       64
#define MAX_DEG   10

static_assert(N_EDGES % 128 == 0, "edge kernels assume full 128-row tiles");
static_assert(N_GRAPHS % 128 == 0, "readout kernels assume full 128-row tiles");

typedef __attribute__((ext_vector_type(16))) _Float16 v16h;
typedef __attribute__((ext_vector_type(8)))  float    v8f;

// ---------------- WMMA helpers (gfx1250, wave32) ----------------

__device__ __forceinline__ v8f wmma32(v16h a, v16h b, v8f c) {
  // D = A(16x32 f16) x B(32x16 f16) + C(16x16 f32)
  return __builtin_amdgcn_wmma_f32_16x16x32_f16(false, a, false, b, (short)0, c,
                                                false, false);
}

__device__ __forceinline__ float sigmoidf(float x) {
  return 1.0f / (1.0f + __expf(-x));
}

// Pack W[K, NOUT] (f32, row-major) into LDS as WMMA B-fragments (f16).
// Fragment (kt, nt) covers K rows [kt*32, kt*32+32), cols [nt*16, nt*16+16).
// Lane-major, 16 contiguous halves per lane -> two ds_load_b128 per B-frag.
// B frag element e of lane L: k = (e&7) + 8*(L>>4) + 16*(e>>3), n = L&15.
__device__ __forceinline__ void pack_w(_Float16* lds, const float* __restrict__ W,
                                       int K, int NOUT, int KT, int NT) {
  const int total = KT * NT * 512;
  for (int idx = threadIdx.x; idx < total; idx += 256) {
    int frag = idx >> 9;
    int rem  = idx & 511;
    int lane = rem >> 4, e = rem & 15;
    int kt = frag / NT, nt = frag - kt * NT;
    int kg = lane >> 4, n = lane & 15;
    int k = kt * 32 + (e & 7) + 8 * kg + 16 * (e >> 3);
    float v = (k < K) ? W[(long long)k * NOUT + nt * 16 + n] : 0.0f;
    lds[idx] = (_Float16)v;
  }
}

__device__ __forceinline__ v16h get_b(const _Float16* lds, int NT, int kt, int nt) {
  const int lane = threadIdx.x & 31;
  const v16h* p = (const v16h*)lds;
  return p[((kt * NT + nt) << 5) + lane];
}

// A-fragment: 16 rows (row = row0 + (lane&15)), K-chunk [k0, k0+32).
// Element e of lane L: k = k0 + 8*(L>>4) + (e&7) + 16*(e>>3).
__device__ __forceinline__ v16h load_a(const float* __restrict__ base,
                                       long long row, int ld, int k0, int K,
                                       bool valid) {
  v16h a = {};
  if (valid) {
    const float* rp = base + row * (long long)ld;
    const int kg = (threadIdx.x >> 4) & 1;
    const int b0 = k0 + 8 * kg;
    const int b1 = b0 + 16;
#pragma unroll
    for (int i = 0; i < 8; ++i) {
      a[i]     = (_Float16)((b0 + i < K) ? rp[b0 + i] : 0.0f);
      a[i + 8] = (_Float16)((b1 + i < K) ? rp[b1 + i] : 0.0f);
    }
  }
  return a;
}

__device__ __forceinline__ void atomicMaxF(float* addr, float val) {
  unsigned int* ua = (unsigned int*)addr;
  unsigned int cur = __float_as_uint(*addr);
  while (__uint_as_float(cur) < val) {
    unsigned int prev = atomicCAS(ua, cur, __float_as_uint(val));
    if (prev == cur) break;
    cur = prev;
  }
}

// ---------------- Generic fused linear: Y = act(X[R,K] @ W[K,64] + b) --------

template <int ACT>
__global__ void __launch_bounds__(256)
wmma_linear_kernel(const float* __restrict__ X, const float* __restrict__ W,
                   const float* __restrict__ Bv, float* __restrict__ Y,
                   int R, int K) {
  __shared__ _Float16 lw[4 * 4 * 512];  // up to K=128, NOUT=64
  __shared__ float lb[EMB];
  const int KT = (K + 31) >> 5;
  pack_w(lw, W, K, EMB, KT, 4);
  for (int i = threadIdx.x; i < EMB; i += 256) lb[i] = Bv[i];
  __syncthreads();

  const int lane = threadIdx.x & 31;
  const int wave = threadIdx.x >> 5;
  const int row0 = blockIdx.x * 128 + wave * 16;
  if (row0 >= R) return;

  v8f acc0 = {}, acc1 = {}, acc2 = {}, acc3 = {};
  const long long arow = row0 + (lane & 15);
  for (int kt = 0; kt < KT; ++kt) {
    v16h a = load_a(X, arow, K, kt * 32, K, arow < R);
    acc0 = wmma32(a, get_b(lw, 4, kt, 0), acc0);
    acc1 = wmma32(a, get_b(lw, 4, kt, 1), acc1);
    acc2 = wmma32(a, get_b(lw, 4, kt, 2), acc2);
    acc3 = wmma32(a, get_b(lw, 4, kt, 3), acc3);
  }
  const int n = lane & 15, hs = lane >> 4;
  v8f accs[4] = {acc0, acc1, acc2, acc3};
  float bv[4];
#pragma unroll
  for (int nt = 0; nt < 4; ++nt) bv[nt] = lb[nt * 16 + n];

  if (row0 + 16 <= R) {  // wave-uniform full tile: straight-line stores
#pragma unroll
    for (int nt = 0; nt < 4; ++nt) {
#pragma unroll
      for (int v = 0; v < 8; ++v) {
        float o = accs[nt][v] + bv[nt];
        if (ACT) o = fmaxf(o, 0.0f);
        Y[(long long)(row0 + v + 8 * hs) * EMB + nt * 16 + n] = o;
      }
    }
  } else {
#pragma unroll
    for (int nt = 0; nt < 4; ++nt) {
#pragma unroll
      for (int v = 0; v < 8; ++v) {
        int r = row0 + v + 8 * hs;
        if (r < R) {
          float o = accs[nt][v] + bv[nt];
          if (ACT) o = fmaxf(o, 0.0f);
          Y[(long long)r * EMB + nt * 16 + n] = o;
        }
      }
    }
  }
}

// ---- MFConv: m[n] = relu(agg[n] @ Wl[deg] + bl[deg] + out[n] @ Wr[deg]) ----
// Branch-free select: each row matches exactly one degree, so selection is
// accumulation with a {0,1} mask -> v_cndmask(const) + v_fmac per element.

__global__ void __launch_bounds__(256)
mfconv_kernel(const float* __restrict__ Agg, const float* __restrict__ Out,
              const int* __restrict__ Deg, const float* __restrict__ Wl,
              const float* __restrict__ Bl, const float* __restrict__ Wr,
              float* __restrict__ M, int R) {
  __shared__ _Float16 lwl[2 * 4 * 512];
  __shared__ _Float16 lwr[2 * 4 * 512];
  __shared__ float lbl[EMB];
  __shared__ int ldeg[128];
  const int tid = threadIdx.x;
  const int blk0 = blockIdx.x * 128;
  for (int i = tid; i < 128; i += 256) {
    int r = blk0 + i;
    int d = (r < R) ? Deg[r] : -1;
    ldeg[i] = (d > MAX_DEG) ? MAX_DEG : d;
  }
  __syncthreads();

  const int lane = tid & 31;
  const int wave = tid >> 5;
  const int row0 = blk0 + wave * 16;
  const long long arow = row0 + (lane & 15);
  const int n = lane & 15, hs = lane >> 4;

  v16h aa[2], oa[2];
#pragma unroll
  for (int kt = 0; kt < 2; ++kt) {
    aa[kt] = load_a(Agg, arow, EMB, kt * 32, EMB, arow < R);
    oa[kt] = load_a(Out, arow, EMB, kt * 32, EMB, arow < R);
  }
  int degv[8];
#pragma unroll
  for (int v = 0; v < 8; ++v) degv[v] = ldeg[wave * 16 + v + 8 * hs];

  v8f mres[4];
#pragma unroll
  for (int nt = 0; nt < 4; ++nt) mres[nt] = (v8f){};

  for (int d = 0; d <= MAX_DEG; ++d) {
    __syncthreads();  // WAR: previous iteration's B-frag reads done
    pack_w(lwl, Wl + (long long)d * EMB * EMB, EMB, EMB, 2, 4);
    pack_w(lwr, Wr + (long long)d * EMB * EMB, EMB, EMB, 2, 4);
    for (int i = tid; i < EMB; i += 256) lbl[i] = Bl[d * EMB + i];
    if (d < MAX_DEG) {  // prefetch next degree's 16KB weight pair
      __builtin_prefetch(Wl + (long long)(d + 1) * EMB * EMB + tid * 16, 0, 3);
      __builtin_prefetch(Wr + (long long)(d + 1) * EMB * EMB + tid * 16, 0, 3);
    }
    __syncthreads();  // RAW: pack visible
    if (row0 < R) {
      float dmask[8];
#pragma unroll
      for (int v = 0; v < 8; ++v) dmask[v] = (degv[v] == d) ? 1.0f : 0.0f;
      float blv[4];
#pragma unroll
      for (int nt = 0; nt < 4; ++nt) blv[nt] = lbl[nt * 16 + n];
#pragma unroll
      for (int nt = 0; nt < 4; ++nt) {
        v8f acc = {};
#pragma unroll
        for (int kt = 0; kt < 2; ++kt) {
          acc = wmma32(aa[kt], get_b(lwl, 4, kt, nt), acc);
          acc = wmma32(oa[kt], get_b(lwr, 4, kt, nt), acc);
        }
#pragma unroll
        for (int v = 0; v < 8; ++v) {
          float o = fmaxf(acc[v] + blv[nt], 0.0f);
          mres[nt][v] = __builtin_fmaf(dmask[v], o, mres[nt][v]);
        }
      }
    }
  }

  if (row0 + 16 <= R) {
#pragma unroll
    for (int nt = 0; nt < 4; ++nt)
#pragma unroll
      for (int v = 0; v < 8; ++v)
        M[(long long)(row0 + v + 8 * hs) * EMB + nt * 16 + n] = mres[nt][v];
  } else if (row0 < R) {
#pragma unroll
    for (int nt = 0; nt < 4; ++nt)
#pragma unroll
      for (int v = 0; v < 8; ++v) {
        int r = row0 + v + 8 * hs;
        if (r < R) M[(long long)r * EMB + nt * 16 + n] = mres[nt][v];
      }
  }
}

// ---- Fused GRU: h = GRU(x=m, h); both 192-wide gate GEMMs + elementwise ----

__global__ void __launch_bounds__(256)
gru_kernel(const float* __restrict__ Xm, float* __restrict__ H,
           const float* __restrict__ Wih, const float* __restrict__ Whh,
           const float* __restrict__ bih, const float* __restrict__ bhh, int R) {
  __shared__ _Float16 lwi[2 * 12 * 512];
  __shared__ _Float16 lwh[2 * 12 * 512];
  __shared__ float lbi[192];
  __shared__ float lbh[192];
  pack_w(lwi, Wih, EMB, 192, 2, 12);
  pack_w(lwh, Whh, EMB, 192, 2, 12);
  for (int i = threadIdx.x; i < 192; i += 256) { lbi[i] = bih[i]; lbh[i] = bhh[i]; }
  __syncthreads();

  const int lane = threadIdx.x & 31;
  const int wave = threadIdx.x >> 5;
  const int row0 = blockIdx.x * 128 + wave * 16;
  if (row0 >= R) return;
  const long long arow = row0 + (lane & 15);
  v16h xa[2], ha[2];
#pragma unroll
  for (int kt = 0; kt < 2; ++kt) {
    xa[kt] = load_a(Xm, arow, EMB, kt * 32, EMB, arow < R);
    ha[kt] = load_a(H,  arow, EMB, kt * 32, EMB, arow < R);
  }
  const int n = lane & 15, hs = lane >> 4;
  const bool full = (row0 + 16 <= R);
#pragma unroll
  for (int nt = 0; nt < 4; ++nt) {
    v8f ir = {}, iz = {}, in_ = {}, hr = {}, hz = {}, hn = {};
#pragma unroll
    for (int kt = 0; kt < 2; ++kt) {
      ir  = wmma32(xa[kt], get_b(lwi, 12, kt, nt),     ir);
      iz  = wmma32(xa[kt], get_b(lwi, 12, kt, nt + 4), iz);
      in_ = wmma32(xa[kt], get_b(lwi, 12, kt, nt + 8), in_);
      hr  = wmma32(ha[kt], get_b(lwh, 12, kt, nt),     hr);
      hz  = wmma32(ha[kt], get_b(lwh, 12, kt, nt + 4), hz);
      hn  = wmma32(ha[kt], get_b(lwh, 12, kt, nt + 8), hn);
    }
    const int c = nt * 16 + n;
    const float bir = lbi[c], biz = lbi[64 + c], bin = lbi[128 + c];
    const float bhr = lbh[c], bhz = lbh[64 + c], bhn = lbh[128 + c];
    if (full) {
#pragma unroll
      for (int v = 0; v < 8; ++v) {
        long long p = (long long)(row0 + v + 8 * hs) * EMB + c;
        float hv = H[p];
        float rg = sigmoidf(ir[v] + bir + hr[v] + bhr);
        float zg = sigmoidf(iz[v] + biz + hz[v] + bhz);
        float ng = tanhf(in_[v] + bin + rg * (hn[v] + bhn));
        H[p] = (1.0f - zg) * ng + zg * hv;
      }
    } else {
#pragma unroll
      for (int v = 0; v < 8; ++v) {
        int r = row0 + v + 8 * hs;
        if (r < R) {
          long long p = (long long)r * EMB + c;
          float hv = H[p];
          float rg = sigmoidf(ir[v] + bir + hr[v] + bhr);
          float zg = sigmoidf(iz[v] + biz + hz[v] + bhz);
          float ng = tanhf(in_[v] + bin + rg * (hn[v] + bhn));
          H[p] = (1.0f - zg) * ng + zg * hv;
        }
      }
    }
  }
}

// ---- EdgeConv: msg = [xi, xj-xi] @ W + b, fused segment-max scatter --------
// N_EDGES % 128 == 0 -> every tile is full; no row guards anywhere.

__global__ void __launch_bounds__(256)
edgeconv_kernel(const float* __restrict__ He, const int* __restrict__ Rw,
                const int* __restrict__ Cl, const float* __restrict__ W,
                const float* __restrict__ Bv, float* __restrict__ AggMax) {
  __shared__ _Float16 lw[4 * 4 * 512];
  __shared__ float lb[EMB];
  __shared__ int lcol[128];
  const int tid = threadIdx.x;
  const int blk0 = blockIdx.x * 128;
  pack_w(lw, W, 128, EMB, 4, 4);
  for (int i = tid; i < EMB; i += 256) lb[i] = Bv[i];
  for (int i = tid; i < 128; i += 256) lcol[i] = Cl[blk0 + i];
  __syncthreads();

  const int lane = tid & 31;
  const int wave = tid >> 5;
  const int e0 = blk0 + wave * 16;
  const int ea = e0 + (lane & 15);
  const float* pi = He + (long long)Cl[ea] * EMB;  // xi = out_e[col]
  const float* pj = He + (long long)Rw[ea] * EMB;  // xj = out_e[row]
  const int kg = lane >> 4;
  v16h afr[4];
#pragma unroll
  for (int kt = 0; kt < 4; ++kt) {
    v16h a;
    const int b0 = kt * 32 + 8 * kg;
#pragma unroll
    for (int i = 0; i < 8; ++i) {
      int k0 = b0 + i, k1 = b0 + 16 + i;
      float f0 = (k0 < EMB) ? pi[k0] : (pj[k0 - EMB] - pi[k0 - EMB]);
      float f1 = (k1 < EMB) ? pi[k1] : (pj[k1 - EMB] - pi[k1 - EMB]);
      a[i]     = (_Float16)f0;
      a[i + 8] = (_Float16)f1;
    }
    afr[kt] = a;
  }
  const int n = lane & 15, hs = lane >> 4;
  int segv[8];
#pragma unroll
  for (int v = 0; v < 8; ++v) segv[v] = lcol[wave * 16 + v + 8 * hs];
#pragma unroll
  for (int nt = 0; nt < 4; ++nt) {
    v8f acc = {};
#pragma unroll
    for (int kt = 0; kt < 4; ++kt)
      acc = wmma32(afr[kt], get_b(lw, 4, kt, nt), acc);
    const float bb = lb[nt * 16 + n];
#pragma unroll
    for (int v = 0; v < 8; ++v)
      atomicMaxF(&AggMax[(long long)segv[v] * EMB + nt * 16 + n], acc[v] + bb);
  }
}

// ---------------- Fills, scatters, small ops --------------------------------

__global__ void fill_f32_kernel(float* p, long long n, float v) {
  long long i = (long long)blockIdx.x * blockDim.x + threadIdx.x;
  if (i < n) p[i] = v;
}
__global__ void fill_i32_kernel(int* p, long long n, int v) {
  long long i = (long long)blockIdx.x * blockDim.x + threadIdx.x;
  if (i < n) p[i] = v;
}
__global__ void deg_kernel(const int* __restrict__ Cl, int* __restrict__ Deg, int E) {
  int e = blockIdx.x * blockDim.x + threadIdx.x;
  if (e < E) atomicAdd(&Deg[Cl[e]], 1);
}
__global__ void agg_scatter_kernel(const float* __restrict__ Out,
                                   const int* __restrict__ Rw,
                                   const int* __restrict__ Cl,
                                   float* __restrict__ Agg, int E) {
  long long idx = (long long)blockIdx.x * blockDim.x + threadIdx.x;
  if (idx < (long long)E * EMB) {
    int e = (int)(idx >> 6), f = (int)(idx & 63);
    atomicAdd(&Agg[(long long)Cl[e] * EMB + f], Out[(long long)Rw[e] * EMB + f]);
  }
}
__global__ void seg_relu_kernel(const float* __restrict__ AggMax,
                                float* __restrict__ Me, long long total,
                                long long nvalid) {
  long long idx = (long long)blockIdx.x * blockDim.x + threadIdx.x;
  if (idx < total)
    Me[idx] = (idx < nvalid) ? fmaxf(AggMax[idx], 0.0f) : 0.0f;
}
__global__ void s_scatter_kernel(const float* __restrict__ Out,
                                 const float* __restrict__ OutE,
                                 const int* __restrict__ Rw,
                                 const int* __restrict__ Cl,
                                 float* __restrict__ S, int E) {
  long long idx = (long long)blockIdx.x * blockDim.x + threadIdx.x;
  if (idx < (long long)E * 128) {
    int e = (int)(idx >> 7), f = (int)(idx & 127);
    float v = (f < 64) ? Out[(long long)Rw[e] * EMB + f]
                       : OutE[(long long)Cl[e] * EMB + (f - 64)];
    atomicAdd(&S[(long long)Cl[e] * 128 + f], v);
  }
}
__global__ void g_scatter_kernel(const float* __restrict__ S,
                                 const int* __restrict__ Batch,
                                 float* __restrict__ G, int N) {
  long long idx = (long long)blockIdx.x * blockDim.x + threadIdx.x;
  if (idx < (long long)N * 128) {
    int nd = (int)(idx >> 7), f = (int)(idx & 127);
    atomicAdd(&G[(long long)Batch[nd] * 128 + f], S[idx]);
  }
}
__global__ void ro2_kernel(const float* __restrict__ G2, const float* __restrict__ W,
                           const float* __restrict__ Bv, float* __restrict__ Y) {
  int r = blockIdx.x * blockDim.x + threadIdx.x;
  if (r < N_GRAPHS) {
    float s = 0.0f;
#pragma unroll
    for (int t = 0; t < 64; ++t) s += G2[r * 64 + t] * W[t];
    Y[r] = s + Bv[0];
  }
}

// ---------------- Host-side orchestration -----------------------------------

extern "C" void kernel_launch(void* const* d_in, const int* in_sizes, int n_in,
                              void* d_out, int out_size, void* d_ws, size_t ws_size,
                              hipStream_t stream) {
  (void)in_sizes; (void)n_in; (void)out_size; (void)ws_size;
  const float* x     = (const float*)d_in[0];
  const float* eattr = (const float*)d_in[1];
  const int*   eidx  = (const int*)d_in[2];
  const int*   batch = (const int*)d_in[3];
  const float* Wa    = (const float*)d_in[4];
  const float* ba    = (const float*)d_in[5];
  const float* Wb    = (const float*)d_in[6];
  const float* bb    = (const float*)d_in[7];
  const float* mfWl  = (const float*)d_in[8];
  const float* mfbl  = (const float*)d_in[9];
  const float* mfWr  = (const float*)d_in[10];
  const float* ecW   = (const float*)d_in[11];
  const float* ecb   = (const float*)d_in[12];
  const float* aWih  = (const float*)d_in[13];
  const float* aWhh  = (const float*)d_in[14];
  const float* abih  = (const float*)d_in[15];
  const float* abhh  = (const float*)d_in[16];
  const float* bWih  = (const float*)d_in[17];
  const float* bWhh  = (const float*)d_in[18];
  const float* bbih  = (const float*)d_in[19];
  const float* bbhh  = (const float*)d_in[20];
  const float* ro0W  = (const float*)d_in[21];
  const float* ro0b  = (const float*)d_in[22];
  const float* ro1W  = (const float*)d_in[23];
  const float* ro1b  = (const float*)d_in[24];
  const float* ro2W  = (const float*)d_in[25];
  const float* ro2b  = (const float*)d_in[26];

  const int* rowi = eidx;
  const int* coli = eidx + N_EDGES;

  float* y = (float*)d_out;
  float* emb_node = y + N_GRAPHS;                          // [N,64] = m buffer
  float* emb_edge = emb_node + (long long)N_NODES * EMB;   // [E,64] = m_e buffer

  char* ws = (char*)d_ws;
  size_t off = 0;
  auto take = [&](size_t bytes) {
    void* p = ws + off;
    off += (bytes + 255) & ~(size_t)255;
    return p;
  };
  int*   deg   = (int*)take(sizeof(int) * (size_t)N_NODES);
  float* hnode = (float*)take(sizeof(float) * (size_t)N_NODES * EMB);
  float* he    = (float*)take(sizeof(float) * (size_t)N_EDGES * EMB);
  float* agg   = (float*)take(sizeof(float) * (size_t)N_NODES * EMB);  // sum & max
  float* sbuf  = (float*)take(sizeof(float) * (size_t)N_NODES * 128);
  float* g     = (float*)take(sizeof(float) * (size_t)N_GRAPHS * 128);
  float* g2    = (float*)take(sizeof(float) * (size_t)N_GRAPHS * EMB);
  float* g3    = (float*)take(sizeof(float) * (size_t)N_GRAPHS * EMB);

  auto nb  = [](long long total) { return (unsigned)((total + 255) / 256); };
  auto nbr = [](long long rows)  { return (unsigned)((rows + 127) / 128); };
  dim3 B(256);

  // degree(col), clamped at use
  fill_i32_kernel<<<nb(N_NODES), B, 0, stream>>>(deg, N_NODES, 0);
  deg_kernel<<<nb(N_EDGES), B, 0, stream>>>(coli, deg, N_EDGES);

  // embeddings (relu)
  wmma_linear_kernel<1><<<nbr(N_NODES), B, 0, stream>>>(x, Wa, ba, hnode, N_NODES, ATOM_DIM);
  wmma_linear_kernel<1><<<nbr(N_EDGES), B, 0, stream>>>(eattr, Wb, bb, he, N_EDGES, BOND_DIM);

  for (int pass = 0; pass < 2; ++pass) {
    // MFConv
    fill_f32_kernel<<<nb((long long)N_NODES * EMB), B, 0, stream>>>(agg, (long long)N_NODES * EMB, 0.0f);
    agg_scatter_kernel<<<nb((long long)N_EDGES * EMB), B, 0, stream>>>(hnode, rowi, coli, agg, N_EDGES);
    mfconv_kernel<<<nbr(N_NODES), B, 0, stream>>>(agg, hnode, deg, mfWl, mfbl, mfWr, emb_node, N_NODES);
    gru_kernel<<<nbr(N_NODES), B, 0, stream>>>(emb_node, hnode, aWih, aWhh, abih, abhh, N_NODES);
    // EdgeConv + segment-max (only first N segments can be non-empty)
    fill_f32_kernel<<<nb((long long)N_NODES * EMB), B, 0, stream>>>(agg, (long long)N_NODES * EMB, -FLT_MAX);
    edgeconv_kernel<<<N_EDGES / 128, B, 0, stream>>>(he, rowi, coli, ecW, ecb, agg);
    seg_relu_kernel<<<nb((long long)N_EDGES * EMB), B, 0, stream>>>(
        agg, emb_edge, (long long)N_EDGES * EMB, (long long)N_NODES * EMB);
    gru_kernel<<<N_EDGES / 128, B, 0, stream>>>(emb_edge, he, bWih, bWhh, bbih, bbhh, N_EDGES);
  }

  // readout: edge->node scatter of concat, node->graph scatter, MLP
  fill_f32_kernel<<<nb((long long)N_NODES * 128), B, 0, stream>>>(sbuf, (long long)N_NODES * 128, 0.0f);
  s_scatter_kernel<<<nb((long long)N_EDGES * 128), B, 0, stream>>>(hnode, he, rowi, coli, sbuf, N_EDGES);
  fill_f32_kernel<<<nb((long long)N_GRAPHS * 128), B, 0, stream>>>(g, (long long)N_GRAPHS * 128, 0.0f);
  g_scatter_kernel<<<nb((long long)N_NODES * 128), B, 0, stream>>>(sbuf, batch, g, N_NODES);
  wmma_linear_kernel<1><<<nbr(N_GRAPHS), B, 0, stream>>>(g, ro0W, ro0b, g2, N_GRAPHS, 128);
  wmma_linear_kernel<1><<<nbr(N_GRAPHS), B, 0, stream>>>(g2, ro1W, ro1b, g3, N_GRAPHS, EMB);
  ro2_kernel<<<nb(N_GRAPHS), B, 0, stream>>>(g3, ro2W, ro2b, y);
}
